// ResidualminGRUblockRL_42743514529994
// MI455X (gfx1250) — compile-verified, matrix operand order checked by reading, not run on hardware
//
#include <hip/hip_runtime.h>
#include <hip/hip_bf16.h>

// ---------------------------------------------------------------------------
// Residual minGRU block for MI455X (gfx1250, wave32, WMMA).
//   B=16, L=4096, N=512.  GEMMs as bf16 WMMA w/ fp32 accum, 2-deep pipelined.
// ---------------------------------------------------------------------------

#define BB   16
#define LL   4096
#define NN   512
#define RR   (BB * LL)          // 65536 rows of u / y
#define EPSV 1e-5f
#define CHK  64                 // scan chunk length
#define NCH  (LL / CHK)         // 64 chunks per sequence

typedef float  v8f   __attribute__((ext_vector_type(8)));
typedef __bf16 v16bf __attribute__((ext_vector_type(16)));

union BfVec {
    uint4 u[2];   // two 16-byte chunks = 16 bf16 = one WMMA A/B operand
    v16bf v;
};

// fp32 -> bf16 with round-to-nearest-even
__device__ __forceinline__ unsigned short f32_to_bf16(float f) {
    unsigned int x = __float_as_uint(f);
    unsigned int r = x + 0x7FFFu + ((x >> 16) & 1u);
    return (unsigned short)(r >> 16);
}

// Load one 16x32 bf16 WMMA operand tile (wave32 A-layout):
//   lanes 0-15  : row = r0+lane,    K = k0+0..7  and k0+16..23
//   lanes 16-31 : row = r0+lane-16, K = k0+8..15 and k0+24..31
// For the B operand, "row" is the W-row (= output column) since
// B[k][n] = W[n][k] and W rows are contiguous in memory.
__device__ __forceinline__ BfVec load_tile16x32(const unsigned short* __restrict__ base,
                                                int row, int k0, int lane) {
    int kb = k0 + ((lane >> 4) << 3);          // +8 for the upper lane half
    const uint4* p = reinterpret_cast<const uint4*>(base + (size_t)row * NN + kb);
    BfVec t;
    t.u[0] = p[0];                              // K = kb .. kb+7
    t.u[1] = p[2];                              // K = kb+16 .. kb+23
    return t;
}

__device__ __forceinline__ float sigmoidf_fast(float x) {
    return 1.0f / (1.0f + __expf(-x));
}

__device__ __forceinline__ float gelu_tanh(float x) {
    float x3 = x * x * x;
    float t  = tanhf(0.7978845608028654f * (x + 0.044715f * x3));
    return 0.5f * x * (1.0f + t);
}

// ---------------------------------------------------------------------------
// fp32 -> bf16 conversion
// ---------------------------------------------------------------------------
__global__ __launch_bounds__(256) void cvt_bf16_kernel(const float* __restrict__ src,
                                                       unsigned short* __restrict__ dst,
                                                       int n) {
    int i = blockIdx.x * blockDim.x + threadIdx.x;
    if (i < n) dst[i] = f32_to_bf16(src[i]);
}

// ---------------------------------------------------------------------------
// Gate GEMMs: accz = u@Wz^T, acch = u@Wh^T  (bf16 WMMA, fp32 accum)
// epilogue:  z = sigmoid(accz+bz);  Acoef = 1-z;  Bcoef = z*(acch+bh)
// 512-thread block (16 waves) per 16-row stripe; each wave owns 2 column
// tiles x 2 matrices.  Operand loads are software-pipelined TWO k-steps
// ahead (wrap-indexed, branch-free) so ~2 full load groups stay in flight
// while the WMMA pipe drains the current step.
// ---------------------------------------------------------------------------
__global__ __launch_bounds__(512) void gates_gemm_kernel(
    const unsigned short* __restrict__ Ubf,
    const unsigned short* __restrict__ Wzb,
    const unsigned short* __restrict__ Whb,
    const float* __restrict__ bz, const float* __restrict__ bh,
    float* __restrict__ Acoef, float* __restrict__ Bcoef)
{
    const int lane = threadIdx.x & 31;
    const int wave = threadIdx.x >> 5;
    const int row0 = blockIdx.x * 16;
    const int colbase = wave * 32;
    const int arow = row0 + (lane & 15);
    const int wr0 = colbase + (lane & 15);
    const int wr1 = colbase + 16 + (lane & 15);

    v8f accz[2], acch[2];
    #pragma unroll
    for (int ct = 0; ct < 2; ++ct) { accz[ct] = v8f{}; acch[ct] = v8f{}; }

    // prologue: fill a 2-deep operand pipeline (k-steps 0 and 1)
    BfVec aP0 = load_tile16x32(Ubf, arow, 0, lane);
    BfVec bzP0[2] = { load_tile16x32(Wzb, wr0, 0, lane), load_tile16x32(Wzb, wr1, 0, lane) };
    BfVec bhP0[2] = { load_tile16x32(Whb, wr0, 0, lane), load_tile16x32(Whb, wr1, 0, lane) };
    BfVec aP1 = load_tile16x32(Ubf, arow, 32, lane);
    BfVec bzP1[2] = { load_tile16x32(Wzb, wr0, 32, lane), load_tile16x32(Wzb, wr1, 32, lane) };
    BfVec bhP1[2] = { load_tile16x32(Whb, wr0, 32, lane), load_tile16x32(Whb, wr1, 32, lane) };

    #pragma unroll
    for (int k0 = 0; k0 < NN; k0 += 32) {
        const int kn = (k0 + 64) & (NN - 1);   // two steps ahead (wraps at end)
        // issue loads for step k0+64 before this step's WMMAs
        BfVec aN = load_tile16x32(Ubf, arow, kn, lane);
        BfVec bzN[2] = { load_tile16x32(Wzb, wr0, kn, lane), load_tile16x32(Wzb, wr1, kn, lane) };
        BfVec bhN[2] = { load_tile16x32(Whb, wr0, kn, lane), load_tile16x32(Whb, wr1, kn, lane) };
        __builtin_prefetch(Ubf + (size_t)arow * NN + kn, 0, 3);   // A stream is HBM-resident

        #pragma unroll
        for (int ct = 0; ct < 2; ++ct) {
            accz[ct] = __builtin_amdgcn_wmma_f32_16x16x32_bf16(
                false, aP0.v, false, bzP0[ct].v, (short)0, accz[ct], false, false);
            acch[ct] = __builtin_amdgcn_wmma_f32_16x16x32_bf16(
                false, aP0.v, false, bhP0[ct].v, (short)0, acch[ct], false, false);
        }
        // rotate the pipeline
        aP0 = aP1; aP1 = aN;
        #pragma unroll
        for (int ct = 0; ct < 2; ++ct) {
            bzP0[ct] = bzP1[ct]; bzP1[ct] = bzN[ct];
            bhP0[ct] = bhP1[ct]; bhP1[ct] = bhN[ct];
        }
    }

    // C/D layout: VGPR i -> (M = (lane>=16 ? 8 : 0)+i, Ncol = lane&15)
    #pragma unroll
    for (int ct = 0; ct < 2; ++ct) {
        int c = colbase + ct * 16 + (lane & 15);
        #pragma unroll
        for (int i = 0; i < 8; ++i) {
            int m = ((lane >> 4) << 3) + i;
            size_t idx = (size_t)(row0 + m) * NN + c;
            float z = sigmoidf_fast(accz[ct][i] + bz[c]);
            float h = acch[ct][i] + bh[c];
            Acoef[idx] = 1.0f - z;
            Bcoef[idx] = z * h;
        }
    }
}

// ---------------------------------------------------------------------------
// Chunked segmented scan, phase 1: compose each 64-step chunk into one
// affine map x -> a*x + b (reset d kills the incoming state: a=0).
// One lane per (b, chunk, n); coalesced over n.
// ---------------------------------------------------------------------------
__global__ __launch_bounds__(256) void scan_chunk_coef_kernel(
    const float* __restrict__ Acoef, const float* __restrict__ Bcoef,
    const unsigned char* __restrict__ dmask,
    float* __restrict__ Ach, float* __restrict__ Bch)
{
    int tid = blockIdx.x * blockDim.x + threadIdx.x;   // 0 .. B*NCH*N-1
    if (tid >= BB * NCH * NN) return;
    int n  = tid & (NN - 1);
    int ch = (tid >> 9) & (NCH - 1);
    int b  = tid >> 15;

    const size_t ebase = ((size_t)b * LL + (size_t)ch * CHK) * NN + n;
    const unsigned char* dm = dmask + (size_t)b * LL + (size_t)ch * CHK;

    float aa = 1.0f, bb = 0.0f;
    for (int t = 0; t < CHK; ++t) {
        size_t idx = ebase + (size_t)t * NN;
        float a  = Acoef[idx];
        float bv = Bcoef[idx];
        if (dm[t]) { aa = 0.0f; bb = bv; }
        else       { aa *= a; bb = fmaf(a, bb, bv); }
    }
    size_t cidx = ((size_t)b * NCH + ch) * NN + n;
    Ach[cidx] = aa;
    Bch[cidx] = bb;
}

// ---------------------------------------------------------------------------
// Scan phase 2: short sequential chain over the 64 chunk maps per channel;
// emits the chunk-entry states and the final x_new output.
// ---------------------------------------------------------------------------
__global__ __launch_bounds__(256) void scan_chain_kernel(
    const float* __restrict__ Ach, const float* __restrict__ Bch,
    const float* __restrict__ x0,
    float* __restrict__ xin, float* __restrict__ xnew_out)
{
    int tid = blockIdx.x * blockDim.x + threadIdx.x;   // 0 .. B*N-1
    if (tid >= BB * NN) return;
    int b = tid >> 9;
    int n = tid & (NN - 1);

    float x = x0[(size_t)b * NN + n];
    for (int ch = 0; ch < NCH; ++ch) {
        size_t cidx = ((size_t)b * NCH + ch) * NN + n;
        xin[cidx] = x;
        x = fmaf(Ach[cidx], x, Bch[cidx]);
    }
    xnew_out[(size_t)b * NN + n] = x;
}

// ---------------------------------------------------------------------------
// Scan phase 3: re-run each chunk from its entry state, fuse GELU, emit bf16.
// ---------------------------------------------------------------------------
__global__ __launch_bounds__(256) void scan_apply_kernel(
    const float* __restrict__ Acoef, const float* __restrict__ Bcoef,
    const unsigned char* __restrict__ dmask,
    const float* __restrict__ xin,
    unsigned short* __restrict__ Ybf)
{
    int tid = blockIdx.x * blockDim.x + threadIdx.x;   // 0 .. B*NCH*N-1
    if (tid >= BB * NCH * NN) return;
    int n  = tid & (NN - 1);
    int ch = (tid >> 9) & (NCH - 1);
    int b  = tid >> 15;

    const size_t ebase = ((size_t)b * LL + (size_t)ch * CHK) * NN + n;
    const unsigned char* dm = dmask + (size_t)b * LL + (size_t)ch * CHK;

    float x = xin[((size_t)b * NCH + ch) * NN + n];
    for (int t = 0; t < CHK; ++t) {
        size_t idx = ebase + (size_t)t * NN;
        float a  = Acoef[idx];
        float bv = Bcoef[idx];
        x = dm[t] ? bv : fmaf(a, x, bv);
        Ybf[idx] = f32_to_bf16(gelu_tanh(x));
    }
}

// ---------------------------------------------------------------------------
// Output GEMMs + gate + residual + LayerNorm.
//   o = (y@W1^T+b1) * sigmoid(y@W2^T+b2) + u ; layernorm over N.
// Same 16-wave, 2-deep-pipelined WMMA structure; pre-norm values staged in
// LDS, two-level LDS reduction for mean/var, then normalize + affine.
// ---------------------------------------------------------------------------
#define LDS_PITCH 513
__global__ __launch_bounds__(512) void out_gemm_ln_kernel(
    const unsigned short* __restrict__ Ybf,
    const unsigned short* __restrict__ W1b,
    const unsigned short* __restrict__ W2b,
    const float* __restrict__ b1, const float* __restrict__ b2,
    const float* __restrict__ u,
    const float* __restrict__ ln_w, const float* __restrict__ ln_b,
    float* __restrict__ out)
{
    __shared__ float lds_val[16 * LDS_PITCH];
    __shared__ float psum[16 * 32];
    __shared__ float psumsq[16 * 32];
    __shared__ float rowmu[16];
    __shared__ float rowrs[16];

    const int lane = threadIdx.x & 31;
    const int wave = threadIdx.x >> 5;
    const int row0 = blockIdx.x * 16;
    const int colbase = wave * 32;
    const int arow = row0 + (lane & 15);
    const int wr0 = colbase + (lane & 15);
    const int wr1 = colbase + 16 + (lane & 15);

    v8f acc1[2], acc2[2];
    #pragma unroll
    for (int ct = 0; ct < 2; ++ct) { acc1[ct] = v8f{}; acc2[ct] = v8f{}; }

    BfVec aP0 = load_tile16x32(Ybf, arow, 0, lane);
    BfVec b1P0[2] = { load_tile16x32(W1b, wr0, 0, lane), load_tile16x32(W1b, wr1, 0, lane) };
    BfVec b2P0[2] = { load_tile16x32(W2b, wr0, 0, lane), load_tile16x32(W2b, wr1, 0, lane) };
    BfVec aP1 = load_tile16x32(Ybf, arow, 32, lane);
    BfVec b1P1[2] = { load_tile16x32(W1b, wr0, 32, lane), load_tile16x32(W1b, wr1, 32, lane) };
    BfVec b2P1[2] = { load_tile16x32(W2b, wr0, 32, lane), load_tile16x32(W2b, wr1, 32, lane) };

    #pragma unroll
    for (int k0 = 0; k0 < NN; k0 += 32) {
        const int kn = (k0 + 64) & (NN - 1);
        BfVec aN = load_tile16x32(Ybf, arow, kn, lane);
        BfVec b1N[2] = { load_tile16x32(W1b, wr0, kn, lane), load_tile16x32(W1b, wr1, kn, lane) };
        BfVec b2N[2] = { load_tile16x32(W2b, wr0, kn, lane), load_tile16x32(W2b, wr1, kn, lane) };
        __builtin_prefetch(Ybf + (size_t)arow * NN + kn, 0, 3);

        #pragma unroll
        for (int ct = 0; ct < 2; ++ct) {
            acc1[ct] = __builtin_amdgcn_wmma_f32_16x16x32_bf16(
                false, aP0.v, false, b1P0[ct].v, (short)0, acc1[ct], false, false);
            acc2[ct] = __builtin_amdgcn_wmma_f32_16x16x32_bf16(
                false, aP0.v, false, b2P0[ct].v, (short)0, acc2[ct], false, false);
        }
        aP0 = aP1; aP1 = aN;
        #pragma unroll
        for (int ct = 0; ct < 2; ++ct) {
            b1P0[ct] = b1P1[ct]; b1P1[ct] = b1N[ct];
            b2P0[ct] = b2P1[ct]; b2P1[ct] = b2N[ct];
        }
    }

    // gate + residual -> LDS (pre-norm values)
    #pragma unroll
    for (int ct = 0; ct < 2; ++ct) {
        int c = colbase + ct * 16 + (lane & 15);
        #pragma unroll
        for (int i = 0; i < 8; ++i) {
            int m = ((lane >> 4) << 3) + i;
            float o1 = acc1[ct][i] + b1[c];
            float o2 = acc2[ct][i] + b2[c];
            float v  = o1 * sigmoidf_fast(o2) + u[(size_t)(row0 + m) * NN + c];
            lds_val[m * LDS_PITCH + c] = v;
        }
    }
    __syncthreads();

    // partial sums: thread (r, seg) reduces 16 columns of row r
    {
        int r   = threadIdx.x >> 5;   // 0..15
        int seg = threadIdx.x & 31;   // 0..31
        float s = 0.0f, sq = 0.0f;
        #pragma unroll
        for (int j = 0; j < 16; ++j) {
            float v = lds_val[r * LDS_PITCH + seg * 16 + j];
            s += v; sq += v * v;
        }
        psum[r * 32 + seg] = s;
        psumsq[r * 32 + seg] = sq;
    }
    __syncthreads();

    if (threadIdx.x < 16) {
        int r = threadIdx.x;
        float s = 0.0f, sq = 0.0f;
        #pragma unroll
        for (int seg = 0; seg < 32; ++seg) { s += psum[r * 32 + seg]; sq += psumsq[r * 32 + seg]; }
        float mu  = s * (1.0f / NN);
        float var = sq * (1.0f / NN) - mu * mu;
        rowmu[r] = mu;
        rowrs[r] = rsqrtf(var + EPSV);
    }
    __syncthreads();

    // normalize + affine + store: each thread handles 16 elements
    #pragma unroll
    for (int e = 0; e < 16; ++e) {
        int flat = threadIdx.x * 16 + e;
        int r = flat >> 9;
        int c = flat & (NN - 1);
        float v = (lds_val[r * LDS_PITCH + c] - rowmu[r]) * rowrs[r] * ln_w[c] + ln_b[c];
        out[(size_t)(row0 + r) * NN + c] = v;
    }
}

// ---------------------------------------------------------------------------
// Host launcher
// ---------------------------------------------------------------------------
extern "C" void kernel_launch(void* const* d_in, const int* in_sizes, int n_in,
                              void* d_out, int out_size, void* d_ws, size_t ws_size,
                              hipStream_t stream) {
    (void)in_sizes; (void)n_in; (void)out_size; (void)ws_size;

    const float*         u    = (const float*)d_in[0];
    const float*         x0   = (const float*)d_in[1];
    const unsigned char* dms  = (const unsigned char*)d_in[2];   // jnp bool -> 1 byte
    const float*         Wz   = (const float*)d_in[3];
    const float*         bz   = (const float*)d_in[4];
    const float*         Wh   = (const float*)d_in[5];
    const float*         bh   = (const float*)d_in[6];
    const float*         W1   = (const float*)d_in[7];
    const float*         b1   = (const float*)d_in[8];
    const float*         W2   = (const float*)d_in[9];
    const float*         b2   = (const float*)d_in[10];
    const float*         lnw  = (const float*)d_in[11];
    const float*         lnb  = (const float*)d_in[12];

    float* y_out  = (float*)d_out;                       // (B, L, N)
    float* x_out  = y_out + (size_t)RR * NN;             // (B, N)

    // workspace carve-up
    unsigned short* Ubf = (unsigned short*)d_ws;                         // R*N bf16
    unsigned short* Wzb = Ubf + (size_t)RR * NN;                          // N*N bf16
    unsigned short* Whb = Wzb + (size_t)NN * NN;
    unsigned short* W1b = Whb + (size_t)NN * NN;
    unsigned short* W2b = W1b + (size_t)NN * NN;
    float* Acoef = (float*)(W2b + (size_t)NN * NN);                       // R*N f32
    float* Bcoef = Acoef + (size_t)RR * NN;                               // R*N f32
    unsigned short* Ybf = (unsigned short*)(Bcoef + (size_t)RR * NN);     // R*N bf16
    float* Ach = (float*)(Ybf + (size_t)RR * NN);                         // B*NCH*N f32
    float* Bch = Ach + (size_t)BB * NCH * NN;
    float* Xin = Bch + (size_t)BB * NCH * NN;

    // 1) bf16 conversions
    {
        int n = RR * NN;
        cvt_bf16_kernel<<<(n + 255) / 256, 256, 0, stream>>>(u, Ubf, n);
        int w = NN * NN;
        cvt_bf16_kernel<<<(w + 255) / 256, 256, 0, stream>>>(Wz, Wzb, w);
        cvt_bf16_kernel<<<(w + 255) / 256, 256, 0, stream>>>(Wh, Whb, w);
        cvt_bf16_kernel<<<(w + 255) / 256, 256, 0, stream>>>(W1, W1b, w);
        cvt_bf16_kernel<<<(w + 255) / 256, 256, 0, stream>>>(W2, W2b, w);
    }

    // 2) gate GEMMs (WMMA, 2-deep pipelined)
    gates_gemm_kernel<<<RR / 16, 512, 0, stream>>>(Ubf, Wzb, Whb, bz, bh, Acoef, Bcoef);

    // 3) chunked segmented scan (coef -> chain -> apply+GELU)
    {
        int nc = BB * NCH * NN;
        scan_chunk_coef_kernel<<<(nc + 255) / 256, 256, 0, stream>>>(Acoef, Bcoef, dms, Ach, Bch);
        scan_chain_kernel<<<(BB * NN + 255) / 256, 256, 0, stream>>>(Ach, Bch, x0, Xin, x_out);
        scan_apply_kernel<<<(nc + 255) / 256, 256, 0, stream>>>(Acoef, Bcoef, dms, Xin, Ybf);
    }

    // 4) output GEMMs + gate + residual + LayerNorm (WMMA, 2-deep pipelined)
    out_gemm_ln_kernel<<<RR / 16, 512, 0, stream>>>(Ybf, W1b, W2b, b1, b2, u, lnw, lnb, y_out);
}
